// rbm_trans_flip_47047071760741
// MI455X (gfx1250) — matrix operand order repaired and checked by reading, not compiled, not run on hardware
//
#include <hip/hip_runtime.h>
#include <hip/hip_bf16.h>

typedef __attribute__((ext_vector_type(16))) __bf16 v16bf;
typedef __attribute__((ext_vector_type(8)))  __bf16 v8bf;
typedef __attribute__((ext_vector_type(8)))  float  v8f;

#define B_DIM 1024
#define N_DIM 1024
#define T_DIM 16
#define H_DIM 4096
#define M_DIM (B_DIM * T_DIM)   // 16384
#define LOG2F_C 0.6931471805599453f

// LDS byte offset of a generic pointer into a __shared__ array
__device__ __forceinline__ unsigned lds_off(const void* p) {
    return (unsigned)(size_t)(const __attribute__((address_space(3))) void*)p;
}

// gfx1250 async LDS-DMA: 16B per lane, tracked by ASYNCcnt
#define ASYNC_LOAD_B128(gptr, lptr)                                          \
    asm volatile("global_load_async_to_lds_b128 %0, %1, off"                 \
                 :: "v"(lds_off(lptr)),                                      \
                    "v"((unsigned long long)(size_t)(gptr)) : "memory")

// complex log_cosh: sgn-flip, z + log1p(exp(-2z)) - log2
__device__ __forceinline__ void log_cosh_c(float zr, float zi, float& orr, float& oii) {
    float s = (zr < 0.f) ? -1.f : 1.f;
    zr *= s; zi *= s;
    float er = __expf(-2.f * zr);
    float wr = er * __cosf(2.f * zi);
    float wi = -er * __sinf(2.f * zi);
    float ar = 1.f + wr;
    orr = zr + 0.5f * __logf(ar * ar + wi * wi) - LOG2F_C;
    oii = zi + atan2f(wi, ar);
}

// xt[m=b*16+t][n] = bf16(x[b][trans[t][n]])
__global__ __launch_bounds__(256) void k_gather_bf16(
    const float* __restrict__ x, const int* __restrict__ tr, __bf16* __restrict__ xt) {
    int i = blockIdx.x * 256 + threadIdx.x;      // 0 .. 16M-1
    int n = i & (N_DIM - 1);
    int m = i >> 10;
    int t = m & (T_DIM - 1);
    int b = m >> 4;
    int src = tr[t * N_DIM + n];
    xt[i] = (__bf16)x[b * N_DIM + src];
}

// WT[h][n] = bf16(W[n][h])   (W is N x H row-major)
__global__ __launch_bounds__(256) void k_wT_bf16(
    const float* __restrict__ W, __bf16* __restrict__ WT) {
    int i = blockIdx.x * 256 + threadIdx.x;      // i = h*1024 + n
    int n = i & (N_DIM - 1);
    int h = i >> 10;
    WT[i] = (__bf16)W[(size_t)n * H_DIM + h];
}

__global__ __launch_bounds__(256) void k_zero(float* __restrict__ p) {
    p[blockIdx.x * 256 + threadIdx.x] = 0.f;
}

// Fused bf16 WMMA GEMM (real+imag) + complex log_cosh + partial H-reduction.
// grid = (H/64, M/128), block = 256 (8 waves). Wave tile: 32x32 per matrix.
// Tiles staged with async LDS-DMA, double-buffered (ASYNCcnt pipelining).
__global__ __launch_bounds__(256) void k_gemm_logcosh(
    const __bf16* __restrict__ xt,    // [M][N]
    const __bf16* __restrict__ WrT,   // [H][N]
    const __bf16* __restrict__ WiT,   // [H][N]
    const float* __restrict__ br, const float* __restrict__ bi,
    float* __restrict__ pos, float* __restrict__ neg)  // 2*M floats each (re,im)
{
    __shared__ __align__(16) __bf16 sA [2][128 * 32];
    __shared__ __align__(16) __bf16 sBr[2][ 64 * 32];
    __shared__ __align__(16) __bf16 sBi[2][ 64 * 32];

    const int tid   = threadIdx.x;
    const int lane  = tid & 31;
    const int wid   = tid >> 5;
    const int wm    = wid & 3;        // 4 row-waves
    const int wh    = wid >> 2;       // 2 col-waves
    const int l15   = lane & 15;
    const int lhalf = lane >> 4;
    const int row0  = blockIdx.y * 128;
    const int col0  = blockIdx.x * 64;

    // per-thread staging coordinates (16B per async op)
    const int idx = tid * 8;
    const int rA  = idx >> 5, cA = idx & 31;   // A tile: 2 passes of 64 rows
    v8f cr[2][2] = {}; v8f ci[2][2] = {};

    // prologue: stage k=0 into buffer 0
    {
        ASYNC_LOAD_B128(&xt [(size_t)(row0 + rA)      * N_DIM + cA], &sA [0][idx]);
        ASYNC_LOAD_B128(&xt [(size_t)(row0 + rA + 64) * N_DIM + cA], &sA [0][idx + 2048]);
        ASYNC_LOAD_B128(&WrT[(size_t)(col0 + rA)      * N_DIM + cA], &sBr[0][idx]);
        ASYNC_LOAD_B128(&WiT[(size_t)(col0 + rA)      * N_DIM + cA], &sBi[0][idx]);
    }

    for (int kk = 0; kk < N_DIM; kk += 32) {
        const int cur = (kk >> 5) & 1;
        const bool has_next = (kk + 32) < N_DIM;
        if (has_next) {
            const int kn = kk + 32, nb = cur ^ 1;
            ASYNC_LOAD_B128(&xt [(size_t)(row0 + rA)      * N_DIM + kn + cA], &sA [nb][idx]);
            ASYNC_LOAD_B128(&xt [(size_t)(row0 + rA + 64) * N_DIM + kn + cA], &sA [nb][idx + 2048]);
            ASYNC_LOAD_B128(&WrT[(size_t)(col0 + rA)      * N_DIM + kn + cA], &sBr[nb][idx]);
            ASYNC_LOAD_B128(&WiT[(size_t)(col0 + rA)      * N_DIM + kn + cA], &sBi[nb][idx]);
            asm volatile("s_wait_asynccnt 0x4" ::: "memory");   // group kk complete
        } else {
            asm volatile("s_wait_asynccnt 0x0" ::: "memory");
        }
        __syncthreads();    // all waves' buffer `cur` resident

        // A fragments: lane half 0 -> K{0-7,16-23}, half 1 -> K{8-15,24-31}
        v16bf a[2], bfr[2], bfi[2];
        #pragma unroll
        for (int ti = 0; ti < 2; ++ti) {
            int r = wm * 32 + ti * 16 + l15;
            int k0 = lhalf * 8;
            v8bf lo = *(const v8bf*)&sA[cur][r * 32 + k0];
            v8bf hi = *(const v8bf*)&sA[cur][r * 32 + k0 + 16];
            #pragma unroll
            for (int e = 0; e < 8; ++e) { a[ti][e] = lo[e]; a[ti][e + 8] = hi[e]; }
        }
        // B fragments: per-lane column, half 0 -> K{0-7,16-23}, half 1 -> K{8-15,24-31}
        #pragma unroll
        for (int tj = 0; tj < 2; ++tj) {
            int h = wh * 32 + tj * 16 + l15;
            int kb = lhalf * 8;
            v8bf lo = *(const v8bf*)&sBr[cur][h * 32 + kb];
            v8bf hi = *(const v8bf*)&sBr[cur][h * 32 + kb + 16];
            #pragma unroll
            for (int e = 0; e < 8; ++e) { bfr[tj][e] = lo[e]; bfr[tj][e + 8] = hi[e]; }
            lo = *(const v8bf*)&sBi[cur][h * 32 + kb];
            hi = *(const v8bf*)&sBi[cur][h * 32 + kb + 16];
            #pragma unroll
            for (int e = 0; e < 8; ++e) { bfi[tj][e] = lo[e]; bfi[tj][e + 8] = hi[e]; }
        }

        #pragma unroll
        for (int ti = 0; ti < 2; ++ti)
            #pragma unroll
            for (int tj = 0; tj < 2; ++tj) {
                cr[ti][tj] = __builtin_amdgcn_wmma_f32_16x16x32_bf16(
                    false, a[ti], false, bfr[tj], (short)0, cr[ti][tj], false, false);
                ci[ti][tj] = __builtin_amdgcn_wmma_f32_16x16x32_bf16(
                    false, a[ti], false, bfi[tj], (short)0, ci[ti][tj], false, false);
            }
        __syncthreads();    // reads of `cur` done -> safe to restage it next iter
    }

    // epilogue: t = cr + i*ci ; accumulate sum_h log_cosh(+-t + b)
    float bre[2], bim[2];
    #pragma unroll
    for (int tj = 0; tj < 2; ++tj) {
        int h = col0 + wh * 32 + tj * 16 + l15;
        bre[tj] = br[h]; bim[tj] = bi[h];
    }
    #pragma unroll
    for (int ti = 0; ti < 2; ++ti) {
        #pragma unroll
        for (int j = 0; j < 8; ++j) {     // C layout: M = base + j + 8*lane_half
            float pr = 0.f, pi = 0.f, nr = 0.f, ni = 0.f;
            #pragma unroll
            for (int tj = 0; tj < 2; ++tj) {
                float tr = cr[ti][tj][j], tim = ci[ti][tj][j];
                float lr, li;
                log_cosh_c( tr + bre[tj],  tim + bim[tj], lr, li); pr += lr; pi += li;
                log_cosh_c(-tr + bre[tj], -tim + bim[tj], lr, li); nr += lr; ni += li;
            }
            // reduce 16 columns within each lane-half
            #pragma unroll
            for (int m = 1; m < 16; m <<= 1) {
                pr += __shfl_xor(pr, m, 32); pi += __shfl_xor(pi, m, 32);
                nr += __shfl_xor(nr, m, 32); ni += __shfl_xor(ni, m, 32);
            }
            if (l15 == 0) {
                int row = row0 + wm * 32 + ti * 16 + j + lhalf * 8;
                atomicAdd(&pos[2 * row],     pr);
                atomicAdd(&pos[2 * row + 1], pi);
                atomicAdd(&neg[2 * row],     nr);
                atomicAdd(&neg[2 * row + 1], ni);
            }
        }
    }
}

// one wave32 per batch row: 32-entry complex logsumexp
__global__ __launch_bounds__(256) void k_lse(
    const float* __restrict__ pos, const float* __restrict__ neg,
    float* __restrict__ out) {
    int tid = threadIdx.x;
    int lane = tid & 31;
    int b = blockIdx.x * 8 + (tid >> 5);
    int t = lane & 15;
    const float* src = (lane < 16) ? pos : neg;
    float zr = src[2 * (b * T_DIM + t)];
    float zi = src[2 * (b * T_DIM + t) + 1];
    float m = zr;
    #pragma unroll
    for (int s = 16; s >= 1; s >>= 1) m = fmaxf(m, __shfl_xor(m, s, 32));
    float e  = __expf(zr - m);
    float sr = e * __cosf(zi);
    float si = e * __sinf(zi);
    #pragma unroll
    for (int s = 16; s >= 1; s >>= 1) { sr += __shfl_xor(sr, s, 32); si += __shfl_xor(si, s, 32); }
    if (lane == 0) {
        out[2 * b]     = 0.5f * __logf(sr * sr + si * si) + m;   // real
        out[2 * b + 1] = atan2f(si, sr);                          // imag
    }
}

extern "C" void kernel_launch(void* const* d_in, const int* in_sizes, int n_in,
                              void* d_out, int out_size, void* d_ws, size_t ws_size,
                              hipStream_t stream) {
    const float* x  = (const float*)d_in[0];
    const float* Wr = (const float*)d_in[1];
    const float* Wi = (const float*)d_in[2];
    const float* br = (const float*)d_in[3];
    const float* bi = (const float*)d_in[4];
    const int*   tr = (const int*)d_in[5];

    char* ws = (char*)d_ws;
    __bf16* xt  = (__bf16*)(ws);                               // 32 MB
    __bf16* WrT = (__bf16*)(ws + (32ull << 20));               //  8 MB
    __bf16* WiT = (__bf16*)(ws + (40ull << 20));               //  8 MB
    float*  pos = (float*)(ws + (48ull << 20));                // 128 KB
    float*  neg = (float*)(ws + (48ull << 20) + 2ull * M_DIM * sizeof(float));
    float*  out = (float*)d_out;

    // 1) gather + convert x
    k_gather_bf16<<<(M_DIM * N_DIM) / 256, 256, 0, stream>>>(x, tr, xt);
    // 2) transpose + convert W
    k_wT_bf16<<<(H_DIM * N_DIM) / 256, 256, 0, stream>>>(Wr, WrT);
    k_wT_bf16<<<(H_DIM * N_DIM) / 256, 256, 0, stream>>>(Wi, WiT);
    // 3) zero accumulators (4*M floats: pos+neg contiguous)
    k_zero<<<(4 * M_DIM) / 256, 256, 0, stream>>>(pos);
    // 4) fused WMMA GEMM + log_cosh + H-reduction (async double-buffered tiles)
    dim3 grid(H_DIM / 64, M_DIM / 128);
    k_gemm_logcosh<<<grid, 256, 0, stream>>>(xt, WrT, WiT, br, bi, pos, neg);
    // 5) 32-way complex logsumexp per batch row
    k_lse<<<B_DIM / 8, 256, 0, stream>>>(pos, neg, out);
}
